// ScaleDotProductAttention_33054068310186
// MI455X (gfx1250) — matrix operand order repaired and checked by reading, not compiled, not run on hardware
//
#include <hip/hip_runtime.h>
#include <hip/hip_bf16.h>
#include <math.h>

// Problem constants (reference: B,H,L,D = 8,16,1024,64)
constexpr int BB = 8;
constexpr int HH = 16;
constexpr int LL = 1024;
constexpr int DD = 64;

constexpr int ROWS    = 32;    // query rows per workgroup
constexpr int THREADS = 128;   // 4 waves (wave32)
constexpr int SCPITCH = 1028;  // padded f32 pitch for score tile (bank spread, 16B-aligned)
constexpr int VTPITCH = 40;    // halves per row of transposed V tile (80B, 16B-aligned)

typedef _Float16 half8  __attribute__((ext_vector_type(8)));
typedef _Float16 v16h_t __attribute__((ext_vector_type(16)));
typedef float    v8f_t  __attribute__((ext_vector_type(8)));

union V16H {
    v16h_t   v;
    half8    h[2];
    _Float16 e[16];
};

static __device__ __forceinline__ v8f_t wmma_f16(const V16H& a, const V16H& b, v8f_t c) {
    return __builtin_amdgcn_wmma_f32_16x16x32_f16(false, a.v, false, b.v, (short)0, c, false, false);
}

__global__ __launch_bounds__(THREADS)
void attn_fused_kernel(const float* __restrict__ q,
                       const float* __restrict__ kmat,
                       const float* __restrict__ vmat,
                       const float* __restrict__ bias,
                       const int*   __restrict__ mask,
                       float* __restrict__ out,
                       float* __restrict__ score) {
    __shared__ __align__(16) _Float16 qs[ROWS * DD];        //  4 KB: Q tile (f16, row-major)
    __shared__ __align__(16) _Float16 ks[64 * DD];          //  8 KB: K chunk (f16, row-major)
    __shared__ __align__(16) _Float16 vt[DD * VTPITCH];     //  5 KB: V chunk transposed (f16)
    __shared__ __align__(16) float    sc[ROWS * SCPITCH];   // ~128.5 KB: score tile (f32)
    __shared__ float red[THREADS];
    __shared__ float rinv[ROWS];

    const int tid  = threadIdx.x;
    const int wave = tid >> 5;
    const int lane = tid & 31;
    const int ln   = lane & 15;   // column index within 16x16 tile
    const int hi   = lane >> 4;   // half-wave selector

    const int nblk  = LL / ROWS;               // 32 row-blocks per (b,h)
    const int bh    = blockIdx.x / nblk;       // fused batch*head
    const int rblk  = blockIdx.x % nblk;
    const int bidx  = bh / HH;                 // batch (bias/mask are [B,1,L,L])
    const int qbase = rblk * ROWS;

    // ---------------- Stage Q tile as f16 ----------------
    const float* qg = q + ((size_t)bh * LL + qbase) * DD;
    for (int i = tid; i < ROWS * DD; i += THREADS) qs[i] = (_Float16)qg[i];
    __syncthreads();

    const int mt  = wave & 1;   // which 16-row half of the 32-row tile
    const int ntb = wave >> 1;  // base column tile; wave handles ntb and ntb+2

    // Preload A fragments (Q rows, K-dim = 64 -> two 32-chunks).
    // A layout (16x32 f16): lane m = ln, element e -> K = (e&7) + hi*8 + (e>>3)*16
    V16H aq0, aq1;
    {
        const int base = (mt * 16 + ln) * DD + hi * 8;
        aq0.h[0] = *(const half8*)&qs[base];
        aq0.h[1] = *(const half8*)&qs[base + 16];
        aq1.h[0] = *(const half8*)&qs[base + 32];
        aq1.h[1] = *(const half8*)&qs[base + 48];
    }

    const float* kg = kmat + (size_t)bh * LL * DD;
    const float* bg = bias + (size_t)bidx * LL * LL;
    const int*   mg = mask + (size_t)bidx * LL * LL;

    // ---------------- S = mask(bias * QK^T/8) into LDS ----------------
    for (int c = 0; c < LL / 64; ++c) {
        for (int i = tid; i < 64 * DD; i += THREADS)
            ks[i] = (_Float16)kg[c * 64 * DD + i];
        // Prefetch next K chunk (16 KB = 128 threads x 128B lines) while computing.
        if (c + 1 < LL / 64)
            __builtin_prefetch(&kg[(c + 1) * 64 * DD + tid * 32], 0, 3);
        __syncthreads();

        #pragma unroll
        for (int t = 0; t < 2; ++t) {
            const int nt = ntb + t * 2;
            // B layout (32x16 f16): lane n = ln, element e -> K = (e&7) + hi*8 + (e>>3)*16.
            // B[k][n] = Kmat[n][k], so lane reads halves of K-row n.
            V16H b0, b1;
            const int base = (nt * 16 + ln) * DD + hi * 8;
            b0.h[0] = *(const half8*)&ks[base];
            b0.h[1] = *(const half8*)&ks[base + 16];
            b1.h[0] = *(const half8*)&ks[base + 32];
            b1.h[1] = *(const half8*)&ks[base + 48];

            v8f_t acc = {};
            acc = wmma_f16(aq0, b0, acc);
            acc = wmma_f16(aq1, b1, acc);

            const int gcol = c * 64 + nt * 16 + ln;
            #pragma unroll
            for (int r = 0; r < 8; ++r) {
                const int lrow = mt * 16 + hi * 8 + r;   // C/D: row = r + hi*8
                const int grow = qbase + lrow;
                const size_t bi = (size_t)grow * LL + gcol;
                // bias/mask are read once per workgroup, HH times across the dispatch:
                // non-temporal keeps them from evicting reusable K/V in L2.
                const float bvl = __builtin_nontemporal_load(&bg[bi]);
                const int   mvl = __builtin_nontemporal_load(&mg[bi]);
                float s = acc[r] * 0.125f * bvl;
                if (mvl == 0) s = -10000.0f;
                sc[lrow * SCPITCH + gcol] = s;
            }
        }
        __syncthreads();
    }

    // ---------------- Row softmax in LDS ----------------
    const int srow = tid >> 2;   // 32 rows
    const int part = tid & 3;    // 4 x 256-col segments per row
    float* rowp = &sc[srow * SCPITCH + part * 256];

    float mloc = -3.0e38f;
    for (int j = 0; j < 256; ++j) mloc = fmaxf(mloc, rowp[j]);
    red[tid] = mloc;
    __syncthreads();
    const float rmax = fmaxf(fmaxf(red[srow * 4 + 0], red[srow * 4 + 1]),
                             fmaxf(red[srow * 4 + 2], red[srow * 4 + 3]));
    float ssum = 0.0f;
    for (int j = 0; j < 256; ++j) {
        float e = __expf(rowp[j] - rmax);
        rowp[j] = e;
        ssum += e;
    }
    __syncthreads();
    red[tid] = ssum;
    __syncthreads();
    if (part == 0) {
        rinv[srow] = 1.0f / (red[srow * 4 + 0] + red[srow * 4 + 1] +
                             red[srow * 4 + 2] + red[srow * 4 + 3]);
    }
    __syncthreads();

    // Normalize in LDS and stream probabilities to global (coalesced, write-once -> NT).
    float* scoreg = score + ((size_t)bh * LL + qbase) * LL;
    for (int i = tid; i < ROWS * LL; i += THREADS) {
        const int r = i >> 10, col = i & 1023;
        const float p = sc[r * SCPITCH + col] * rinv[r];
        sc[r * SCPITCH + col] = p;
        __builtin_nontemporal_store(p, &scoreg[(size_t)r * LL + col]);
    }
    __syncthreads();

    // ---------------- out = P @ V ----------------
    const float* vg = vmat + (size_t)bh * LL * DD;
    v8f_t o0 = {}, o1 = {};
    for (int kc = 0; kc < LL / 32; ++kc) {
        // Stage V chunk transposed: vt[d][k] (f16), coalesced global reads.
        for (int i = tid; i < 32 * DD; i += THREADS) {
            const int kr = i >> 6, dc = i & 63;
            vt[dc * VTPITCH + kr] = (_Float16)vg[(size_t)(kc * 32 + kr) * DD + dc];
        }
        // Prefetch next V chunk (8 KB: first 64 threads cover 64 x 128B lines).
        if (kc + 1 < LL / 32 && tid < 64)
            __builtin_prefetch(&vg[(size_t)(kc + 1) * 32 * DD + tid * 32], 0, 3);
        __syncthreads();

        // A fragment: probabilities from LDS score tile (f32 -> f16).
        V16H ap;
        {
            const float* fp = &sc[(mt * 16 + ln) * SCPITCH + kc * 32 + hi * 8];
            #pragma unroll
            for (int e2 = 0; e2 < 8; ++e2) ap.e[e2] = (_Float16)fp[e2];
            #pragma unroll
            for (int e2 = 0; e2 < 8; ++e2) ap.e[8 + e2] = (_Float16)fp[16 + e2];
        }
        // B fragments: column n of V = vt row n; K = (e&7) + hi*8 + (e>>3)*16.
        V16H bv0, bv1;
        const int n0 = (ntb * 16 + ln) * VTPITCH + hi * 8;
        const int n1 = ((ntb + 2) * 16 + ln) * VTPITCH + hi * 8;
        bv0.h[0] = *(const half8*)&vt[n0];
        bv0.h[1] = *(const half8*)&vt[n0 + 16];
        bv1.h[0] = *(const half8*)&vt[n1];
        bv1.h[1] = *(const half8*)&vt[n1 + 16];

        o0 = wmma_f16(ap, bv0, o0);
        o1 = wmma_f16(ap, bv1, o1);
        __syncthreads();
    }

    float* og = out + ((size_t)bh * LL + qbase) * DD;
    #pragma unroll
    for (int r = 0; r < 8; ++r) {
        const int lrow = mt * 16 + hi * 8 + r;
        __builtin_nontemporal_store(o0[r], &og[(size_t)lrow * DD + ntb * 16 + ln]);
        __builtin_nontemporal_store(o1[r], &og[(size_t)lrow * DD + (ntb + 2) * 16 + ln]);
    }
}

extern "C" void kernel_launch(void* const* d_in, const int* in_sizes, int n_in,
                              void* d_out, int out_size, void* d_ws, size_t ws_size,
                              hipStream_t stream) {
    (void)in_sizes; (void)n_in; (void)d_ws; (void)ws_size; (void)out_size;
    const float* q    = (const float*)d_in[0];
    const float* k    = (const float*)d_in[1];
    const float* v    = (const float*)d_in[2];
    const float* bias = (const float*)d_in[3];
    const int*   mask = (const int*)d_in[4];

    float* out   = (float*)d_out;                              // [B,H,L,D]
    float* score = out + (size_t)BB * HH * LL * DD;            // [B,H,L,L]

    const int grid = BB * HH * (LL / ROWS);                    // 4096 workgroups
    attn_fused_kernel<<<grid, THREADS, 0, stream>>>(q, k, v, bias, mask, out, score);
}
